// Attention_9740985827390
// MI455X (gfx1250) — matrix operand order repaired and checked by reading, not compiled, not run on hardware
//
#include <hip/hip_runtime.h>

// ---------------------------------------------------------------------------
// Shapes (compile-time, from the reference)
// ---------------------------------------------------------------------------
#define Bsz   2
#define Nseq  2048
#define DIM   1024
#define Hh    16
#define DH    64
#define NNK   2
#define JP    2080           // padded key length: [0,14)=pad, 14..15=null kv, 16..2063=real, rest pad
#define JNULL0 14
#define JREAL0 16
#define JREAL_END 2064       // 16 + 2048

typedef __attribute__((ext_vector_type(16))) _Float16 v16h;
typedef __attribute__((ext_vector_type(8)))  _Float16 v8h;
typedef __attribute__((ext_vector_type(8)))  float    v8f;

static __device__ __forceinline__ v8f wmma_f16(v16h a, v16h b, v8f c) {
    return __builtin_amdgcn_wmma_f32_16x16x32_f16(false, a, false, b, (short)0, c, false, false);
}

static __device__ __forceinline__ v16h ldA(const _Float16* lo, const _Float16* hi) {
    v8h l = *(const v8h*)lo;
    v8h h = *(const v8h*)hi;
    return __builtin_shufflevector(l, h, 0,1,2,3,4,5,6,7,8,9,10,11,12,13,14,15);
}
static __device__ __forceinline__ v16h ld16(const _Float16* p) { return ldA(p, p + 8); }

// Async DMA: 16 bytes per lane, global -> LDS, tracked by ASYNCcnt.
// ldsaddr = low 32 bits of a generic pointer into __shared__ (== LDS byte offset).
static __device__ __forceinline__ void async_b128(const _Float16* lds_dst, const _Float16* gsrc) {
    const unsigned       ld = (unsigned)(unsigned long long)lds_dst;
    const unsigned long long ga = (unsigned long long)gsrc;
    asm volatile("global_load_async_to_lds_b128 %0, %1, off" :: "v"(ld), "v"(ga) : "memory");
}
static __device__ __forceinline__ void wait_async0() {
    asm volatile("s_wait_asynccnt 0x0" ::: "memory");
}

// ---------------------------------------------------------------------------
// Weight transpose + cast to f16: W[K][Nc] -> Wt[Nc][K]
// ---------------------------------------------------------------------------
__global__ void __launch_bounds__(256) transpose_cast_kernel(const float* __restrict__ W,
                                                             _Float16* __restrict__ Wt,
                                                             int K, int Nc) {
    size_t tid = (size_t)blockIdx.x * 256 + threadIdx.x;
    if (tid >= (size_t)K * Nc) return;
    int o = (int)(tid / K);
    int k = (int)(tid % K);
    Wt[tid] = (_Float16)W[(size_t)k * Nc + o];
}

// ---------------------------------------------------------------------------
// LayerNorm(x)*gamma -> xn16 ; raw x -> x16 (KV input is pre-norm x)
// ---------------------------------------------------------------------------
__global__ void __launch_bounds__(256) ln_cast_kernel(const float* __restrict__ x,
                                                      const float* __restrict__ gamma,
                                                      _Float16* __restrict__ xn16,
                                                      _Float16* __restrict__ x16) {
    __shared__ float red[256];
    __shared__ float red2[256];
    const int row = blockIdx.x;
    const float* xr = x + (size_t)row * DIM;
    float v[4]; float s = 0.f, s2 = 0.f;
    for (int u = 0; u < 4; ++u) {
        v[u] = xr[threadIdx.x + 256 * u];
        s += v[u]; s2 += v[u] * v[u];
    }
    red[threadIdx.x] = s; red2[threadIdx.x] = s2;
    __syncthreads();
    for (int o = 128; o > 0; o >>= 1) {
        if ((int)threadIdx.x < o) {
            red[threadIdx.x]  += red[threadIdx.x + o];
            red2[threadIdx.x] += red2[threadIdx.x + o];
        }
        __syncthreads();
    }
    const float mu  = red[0] * (1.0f / DIM);
    const float var = red2[0] * (1.0f / DIM) - mu * mu;
    const float rs  = rsqrtf(var + 1e-5f);
    for (int u = 0; u < 4; ++u) {
        int c = threadIdx.x + 256 * u;
        xn16[(size_t)row * DIM + c] = (_Float16)((v[u] - mu) * rs * gamma[c]);
        x16 [(size_t)row * DIM + c] = (_Float16)v[u];
    }
}

// ---------------------------------------------------------------------------
// Generic f16 WMMA GEMM:  C[M][N] (f32) = A[M][K] (f16) x Bt[N][K]^T (f16)
// 64x64 tile per block, 8 waves, k-step 32. Double-buffered LDS fed by
// async global->LDS DMA (ASYNCcnt), software pipelined.
// ---------------------------------------------------------------------------
__global__ void __launch_bounds__(256) gemm_f16_tn(const _Float16* __restrict__ A,
                                                   const _Float16* __restrict__ Bt,
                                                   float* __restrict__ C,
                                                   int M, int N, int K) {
    __shared__ _Float16 As[2][64 * 32];
    __shared__ _Float16 Bs[2][64 * 32];
    const int t = threadIdx.x;
    const int l = t & 31;
    const int w = t >> 5;
    const int M0 = blockIdx.y * 64, N0 = blockIdx.x * 64;
    const int mw = (w & 3) * 16, nw = (w >> 2) * 32;
    const int lm = l & 15, kh = l >> 4;
    const int sr = (t * 8) / 32, sc = (t * 8) % 32;   // staging coords (64x32 halves)
    v8f acc0 = {}, acc1 = {};

    const _Float16* gA = &A [(size_t)(M0 + sr) * K + sc];
    const _Float16* gB = &Bt[(size_t)(N0 + sr) * K + sc];

    // prologue: DMA first k-tile into buffer 0
    async_b128(&As[0][sr * 32 + sc], gA);
    async_b128(&Bs[0][sr * 32 + sc], gB);

    int buf = 0;
    for (int kt = 0; kt < K; kt += 32) {
        wait_async0();          // this wave's DMAs for tile `buf` are done
        __syncthreads();        // everyone's are done; prev buffer reads also done
        if (kt + 32 < K) {      // pipeline: DMA next tile into the other buffer
            async_b128(&As[buf ^ 1][sr * 32 + sc], gA + kt + 32);
            async_b128(&Bs[buf ^ 1][sr * 32 + sc], gB + kt + 32);
        }
        if (kt + 64 < K) {      // keep HBM lines in flight one tile further out
            __builtin_prefetch(gA + kt + 64, 0, 3);
            __builtin_prefetch(gB + kt + 64, 0, 3);
        }
        const int m = mw + lm;
        v16h a  = ld16(&As[buf][m * 32 + kh * 16]);
        v16h b0 = ld16(&Bs[buf][(nw + lm)      * 32 + kh * 16]);
        v16h b1 = ld16(&Bs[buf][(nw + 16 + lm) * 32 + kh * 16]);
        acc0 = wmma_f16(a, b0, acc0);
        acc1 = wmma_f16(a, b1, acc1);
        buf ^= 1;
    }
    const int mrow = M0 + mw + 8 * kh;
    const int ncol = N0 + nw + lm;
    for (int v = 0; v < 8; ++v) {
        C[(size_t)(mrow + v) * N + ncol]      = acc0[v];
        C[(size_t)(mrow + v) * N + ncol + 16] = acc1[v];
    }
}

// ---------------------------------------------------------------------------
// q: l2norm * q_scale * SCALE, repack [b,n,h,d] (f32) -> [b,h,n,d] (f16)
// ---------------------------------------------------------------------------
__global__ void __launch_bounds__(64) q_norm_kernel(const float* __restrict__ q32,
                                                    const float* __restrict__ qscale,
                                                    _Float16* __restrict__ q16) {
    __shared__ float red[64];
    const int r  = blockIdx.x;          // (b*N + n)*H + h
    const int h  = r & (Hh - 1);
    const int bn = r >> 4;
    const int d  = threadIdx.x;
    const float val = q32[(size_t)bn * (Hh * DH) + h * DH + d];
    red[d] = val * val;
    __syncthreads();
    for (int o = 32; o > 0; o >>= 1) { if (d < o) red[d] += red[d + o]; __syncthreads(); }
    const float inv = 1.0f / fmaxf(sqrtf(red[0]), 1e-12f);
    const int b = bn >> 11, n = bn & (Nseq - 1);
    q16[(((size_t)(b * Hh + h) * Nseq) + n) * DH + d] = (_Float16)(val * inv * qscale[d] * 8.0f);
}

// ---------------------------------------------------------------------------
// Build padded K/V [b,h,JP,64] f16: zeros pad, null kv at j=14..15, real j>=16.
// ---------------------------------------------------------------------------
__global__ void __launch_bounds__(64) kv_build_kernel(const float* __restrict__ kv32,
                                                      const float* __restrict__ nullkv,
                                                      const float* __restrict__ kscale,
                                                      _Float16* __restrict__ k16,
                                                      _Float16* __restrict__ v16o) {
    __shared__ float red[64];
    const int r  = blockIdx.x;                 // (b*H + h)*JP + jg
    const int jg = r % JP;
    const int bh = r / JP;
    const int h  = bh & (Hh - 1);
    const int b  = bh >> 4;
    const int d  = threadIdx.x;
    float kraw = 0.f, vraw = 0.f; bool valid = false;
    if (jg >= JNULL0 && jg < JREAL0) {
        const int ridx = jg - JNULL0;
        kraw = nullkv[((size_t)h * (2 * NNK) + 2 * ridx)     * DH + d];
        vraw = nullkv[((size_t)h * (2 * NNK) + 2 * ridx + 1) * DH + d];
        valid = true;
    } else if (jg >= JREAL0 && jg < JREAL_END) {
        const int n = jg - JREAL0;
        kraw = kv32[((size_t)(b * Nseq + n)) * (2 * Hh * DH) + h * DH + d];
        vraw = kv32[((size_t)(b * Nseq + n)) * (2 * Hh * DH) + Hh * DH + h * DH + d];
        valid = true;
    }
    red[d] = kraw * kraw;
    __syncthreads();
    for (int o = 32; o > 0; o >>= 1) { if (d < o) red[d] += red[d + o]; __syncthreads(); }
    const float inv = valid ? (1.0f / fmaxf(sqrtf(red[0]), 1e-12f)) : 0.f;
    const size_t base = (((size_t)bh) * JP + jg) * DH + d;
    k16 [base] = (_Float16)(kraw * inv * kscale[d]);
    v16o[base] = (_Float16)vraw;
}

// ---------------------------------------------------------------------------
// Flash attention: block = (b,h, 128 rows), 8 waves x 16-row tiles.
// K tile staged by async global->LDS DMA; V tile transposed through VGPRs
// (overlapping the in-flight DMA); next-tile K/V prefetched. Bias streams
// into the WMMA C operand. fp32 online softmax via 16-lane shuffles.
// ---------------------------------------------------------------------------
__global__ void __launch_bounds__(256) flash_attn_kernel(const _Float16* __restrict__ q16,
                                                         const _Float16* __restrict__ k16,
                                                         const _Float16* __restrict__ v16,
                                                         const float* __restrict__ bias,
                                                         const unsigned char* __restrict__ mask,
                                                         _Float16* __restrict__ attn16) {
    __shared__ _Float16 Kt[32 * 64];      // [j][dh]
    __shared__ _Float16 Vt[64 * 32];      // [dh][j] (transposed)
    __shared__ _Float16 Pb[8][16 * 32];   // per-wave P staging

    const int t = threadIdx.x, l = t & 31, w = t >> 5;
    const int b = blockIdx.z, h = blockIdx.y;
    const int i0  = blockIdx.x * 128;
    const int i0w = i0 + w * 16;
    const int lm = l & 15, kh = l >> 4;

    // Q fragments for this wave's 16 rows (scale/SCALE pre-folded)
    const _Float16* qbase = q16 + (((size_t)(b * Hh + h) * Nseq) + i0w + lm) * DH;
    const v16h qa0 = ld16(qbase +      kh * 16);
    const v16h qa1 = ld16(qbase + 32 + kh * 16);

    v8f o0 = {}, o1 = {}, o2 = {}, o3 = {};
    float mrow[8], lrow[8];
    for (int v = 0; v < 8; ++v) { mrow[v] = -1e30f; lrow[v] = 0.f; }

    const _Float16* kbh = k16 + (size_t)(b * Hh + h) * JP * DH;
    const _Float16* vbh = v16 + (size_t)(b * Hh + h) * JP * DH;
    const unsigned char* mb = mask + (size_t)b * Nseq;
    const float* biash = bias + (size_t)h * Nseq * Nseq;

    const int nsteps = (i0 + 143) / 32 + 1;            // covers jg <= i_max + 16
    const int sr = (t * 8) / 64, sc = (t * 8) % 64;    // staging coords (32x64 halves)
    _Float16* pme = &Pb[w][0];

    for (int s = 0; s < nsteps; ++s) {
        const int j0 = s * 32;
        __syncthreads();
        // async-DMA K tile straight into LDS (no VGPR round trip) ...
        async_b128(&Kt[sr * 64 + sc], &kbh[((size_t)(j0 + sr)) * DH + sc]);
        // ... while V is transposed through VGPRs (overlaps the DMA)
        {
            v8h vv = *(const v8h*)&vbh[((size_t)(j0 + sr)) * DH + sc];
            for (int u = 0; u < 8; ++u) Vt[(sc + u) * 32 + sr] = vv[u];
        }
        if (s + 1 < nsteps) {
            __builtin_prefetch(&kbh[((size_t)(j0 + 32 + sr)) * DH + sc], 0, 3);
            __builtin_prefetch(&vbh[((size_t)(j0 + 32 + sr)) * DH + sc], 0, 3);
        }
        wait_async0();
        __syncthreads();

        v8f sA[2];
        for (int jh = 0; jh < 2; ++jh) {
            const int jg = j0 + jh * 16 + lm;
            // bias streamed straight into the WMMA C operand
            v8f cbias;
            for (int v = 0; v < 8; ++v) {
                const int i = i0w + v + 8 * kh;
                float bb = 0.f;
                if (jg >= JREAL0 && jg < JREAL_END)
                    bb = biash[(size_t)i * Nseq + (jg - JREAL0)];
                cbias[v] = bb;
            }
            v16h kb0 = ld16(&Kt[(jh * 16 + lm) * 64 +      kh * 16]);
            v16h kb1 = ld16(&Kt[(jh * 16 + lm) * 64 + 32 + kh * 16]);
            v8f sv = wmma_f16(qa0, kb0, cbias);
            sv = wmma_f16(qa1, kb1, sv);
            // masking: pad cols, key-padding mask, causal (null kv always ok)
            const bool colpad = (jg < JNULL0) || (jg >= JREAL_END);
            const bool isnull = (jg >= JNULL0) && (jg < JREAL0);
            const int  kn     = jg - JREAL0;
            const bool mk = (!colpad && !isnull) ? (mb[kn < 0 ? 0 : kn] != 0) : true;
            for (int v = 0; v < 8; ++v) {
                const int i = i0w + v + 8 * kh;
                const bool ok = !colpad && (isnull || (mk && kn <= i));
                if (!ok) sv[v] = -1e30f;
            }
            sA[jh] = sv;
        }

        // fp32 online softmax; row m=v (lanes 0-15) / m=v+8 (lanes 16-31)
        for (int v = 0; v < 8; ++v) {
            float tm = fmaxf(sA[0][v], sA[1][v]);
            for (int off = 1; off < 16; off <<= 1) tm = fmaxf(tm, __shfl_xor(tm, off, 32));
            const float mnew  = fmaxf(mrow[v], tm);
            const float alpha = __expf(mrow[v] - mnew);
            mrow[v] = mnew;
            const float p0 = __expf(sA[0][v] - mnew);
            const float p1 = __expf(sA[1][v] - mnew);
            sA[0][v] = p0; sA[1][v] = p1;
            float rs = p0 + p1;
            for (int off = 1; off < 16; off <<= 1) rs += __shfl_xor(rs, off, 32);
            lrow[v] = lrow[v] * alpha + rs;
            o0[v] *= alpha; o1[v] *= alpha; o2[v] *= alpha; o3[v] *= alpha;
        }

        // P: C-layout -> A-layout via per-wave LDS staging
        for (int jh = 0; jh < 2; ++jh)
            for (int v = 0; v < 8; ++v)
                pme[(v + 8 * kh) * 32 + jh * 16 + lm] = (_Float16)sA[jh][v];
        __syncthreads();
        v16h pa  = ld16(&pme[lm * 32 + kh * 16]);
        v16h vb0 = ld16(&Vt[( 0 + lm) * 32 + kh * 16]);
        v16h vb1 = ld16(&Vt[(16 + lm) * 32 + kh * 16]);
        v16h vb2 = ld16(&Vt[(32 + lm) * 32 + kh * 16]);
        v16h vb3 = ld16(&Vt[(48 + lm) * 32 + kh * 16]);
        o0 = wmma_f16(pa, vb0, o0);
        o1 = wmma_f16(pa, vb1, o1);
        o2 = wmma_f16(pa, vb2, o2);
        o3 = wmma_f16(pa, vb3, o3);
    }

    // normalize and store [b*N + i][h*64 + dh] as f16 for the Wo GEMM
    for (int v = 0; v < 8; ++v) {
        const float inv = 1.0f / fmaxf(lrow[v], 1e-20f);
        const int i = i0w + v + 8 * kh;
        const size_t rowb = ((size_t)(b * Nseq) + i) * (Hh * DH) + h * DH;
        attn16[rowb +  0 + lm] = (_Float16)(o0[v] * inv);
        attn16[rowb + 16 + lm] = (_Float16)(o1[v] * inv);
        attn16[rowb + 32 + lm] = (_Float16)(o2[v] * inv);
        attn16[rowb + 48 + lm] = (_Float16)(o3[v] * inv);
    }
}

// ---------------------------------------------------------------------------
// Host launcher
// ---------------------------------------------------------------------------
extern "C" void kernel_launch(void* const* d_in, const int* in_sizes, int n_in,
                              void* d_out, int out_size, void* d_ws, size_t ws_size,
                              hipStream_t stream) {
    (void)in_sizes; (void)n_in; (void)out_size; (void)ws_size;
    const float*         x      = (const float*)d_in[0];
    const unsigned char* mask   = (const unsigned char*)d_in[1];
    const float*         bias   = (const float*)d_in[2];
    const float*         gamma  = (const float*)d_in[3];
    const float*         nullkv = (const float*)d_in[4];
    const float*         Wq     = (const float*)d_in[5];
    const float*         Wkv    = (const float*)d_in[6];
    const float*         qscale = (const float*)d_in[7];
    const float*         kscale = (const float*)d_in[8];
    const float*         Wo     = (const float*)d_in[9];
    float* out = (float*)d_out;

    char* ws = (char*)d_ws; size_t off = 0;
    auto carve = [&](size_t bytes) -> char* {
        char* p = ws + off; off += (bytes + 255) & ~(size_t)255; return p;
    };
    const size_t MT = (size_t)Bsz * Nseq;               // 4096 token rows
    _Float16* xn16 = (_Float16*)carve(MT * DIM * 2);
    _Float16* x16  = (_Float16*)carve(MT * DIM * 2);
    _Float16* WqT  = (_Float16*)carve((size_t)DIM * (Hh * DH) * 2);
    _Float16* WkvT = (_Float16*)carve((size_t)DIM * (2 * Hh * DH) * 2);
    _Float16* WoT  = (_Float16*)carve((size_t)(Hh * DH) * DIM * 2);
    float*    q32  = (float*)carve(MT * (Hh * DH) * 4);
    float*    kv32 = (float*)carve(MT * (2 * Hh * DH) * 4);
    _Float16* q16  = (_Float16*)carve((size_t)Bsz * Hh * Nseq * DH * 2);
    _Float16* k16  = (_Float16*)carve((size_t)Bsz * Hh * JP * DH * 2);
    _Float16* v16b = (_Float16*)carve((size_t)Bsz * Hh * JP * DH * 2);
    _Float16* a16  = (_Float16*)carve(MT * (Hh * DH) * 2);

    // 1) weights -> f16 transposed
    transpose_cast_kernel<<<(DIM * Hh * DH + 255) / 256, 256, 0, stream>>>(Wq,  WqT,  DIM, Hh * DH);
    transpose_cast_kernel<<<(DIM * 2 * Hh * DH + 255) / 256, 256, 0, stream>>>(Wkv, WkvT, DIM, 2 * Hh * DH);
    transpose_cast_kernel<<<(Hh * DH * DIM + 255) / 256, 256, 0, stream>>>(Wo,  WoT,  Hh * DH, DIM);
    // 2) layernorm + casts
    ln_cast_kernel<<<(int)MT, 256, 0, stream>>>(x, gamma, xn16, x16);
    // 3) projections (WMMA GEMMs, async double-buffered)
    dim3 gq(DIM / 64, (int)MT / 64);
    gemm_f16_tn<<<gq, 256, 0, stream>>>(xn16, WqT, q32, (int)MT, DIM, DIM);
    dim3 gkv(2 * DIM / 64, (int)MT / 64);
    gemm_f16_tn<<<gkv, 256, 0, stream>>>(x16, WkvT, kv32, (int)MT, 2 * DIM, DIM);
    // 4) qk norms + padded K/V build
    q_norm_kernel<<<(int)MT * Hh, 64, 0, stream>>>(q32, qscale, q16);
    kv_build_kernel<<<Bsz * Hh * JP, 64, 0, stream>>>(kv32, nullkv, kscale, k16, v16b);
    // 5) flash attention
    dim3 ga(Nseq / 128, Hh, Bsz);
    flash_attn_kernel<<<ga, 256, 0, stream>>>(q16, k16, v16b, bias, mask, a16);
    // 6) output projection
    gemm_f16_tn<<<gq, 256, 0, stream>>>(a16, WoT, out, (int)MT, DIM, DIM);
}
